// RoPEAttention_48576080118373
// MI455X (gfx1250) — compile-verified, matrix operand order checked by reading, not run on hardware
//
#include <hip/hip_runtime.h>

// ---------------------------------------------------------------------------
// RoPE multi-head attention for MI455X (gfx1250, wave32, WMMA bf16).
// B=4, T=2048, D_MODEL=1024, H=16, HEAD_DIM=64.
// ---------------------------------------------------------------------------

typedef __bf16 bf16_t;
typedef __attribute__((ext_vector_type(16))) __bf16 v16bf;
typedef __attribute__((ext_vector_type(8)))  __bf16 v8bf;
typedef __attribute__((ext_vector_type(8)))  float  v8f;

#define D_MODEL 1024
#define NHEADS  16
#define HDIM    64
#define SEQ     2048
#define BATCH   4
#define ROWS    (BATCH * SEQ)      // 8192
#define LOG2E   1.44269504088896340736f

// ---------------------------------------------------------------------------
// Load a 16x32 bf16 tile from a row-major matrix into WMMA A-operand layout.
// Per ISA 7.12.2 (16-bit A 16x32): lane L holds row M=L%16, K chunks at
// base (L<16 ? 0 : 8) and base+16, 8 halves each.  The same routine loads a
// B operand when given B^T (i.e. "row" = output column n).
// ---------------------------------------------------------------------------
static __device__ inline v16bf load_tile(const bf16_t* __restrict__ base,
                                         int ld, int lane) {
    const int r  = lane & 15;
    const int cb = (lane < 16) ? 0 : 8;
    const v8bf* p0 = reinterpret_cast<const v8bf*>(base + (size_t)r * ld + cb);
    const v8bf* p1 = reinterpret_cast<const v8bf*>(base + (size_t)r * ld + cb + 16);
    v8bf lo = *p0;
    v8bf hi = *p1;
    return __builtin_shufflevector(lo, hi, 0, 1, 2, 3, 4, 5, 6, 7,
                                   8, 9, 10, 11, 12, 13, 14, 15);
}

static __device__ inline v8f wmma_bf16(v16bf a, v16bf b, v8f c) {
    return __builtin_amdgcn_wmma_f32_16x16x32_bf16(
        /*neg_a=*/false, a, /*neg_b=*/false, b,
        /*c_mod=*/(short)0, c, /*reuse_a=*/false, /*reuse_b=*/false);
}

static __device__ inline v8f zero8() {
    v8f z;
#pragma unroll
    for (int i = 0; i < 8; ++i) z[i] = 0.0f;
    return z;
}

// ---------------------------------------------------------------------------
// fp32 -> bf16 conversion
// ---------------------------------------------------------------------------
__global__ void cvt_f32_bf16(const float* __restrict__ in,
                             bf16_t* __restrict__ out, int n) {
    int i = blockIdx.x * blockDim.x + threadIdx.x;
    if (i < n) out[i] = (bf16_t)in[i];
}

// ---------------------------------------------------------------------------
// GEMM: out[row, n] = sum_k X[row, k] * W[n, k]   (torch Linear: x @ W^T)
// One wave per block; wave computes 64 rows x 64 cols (16 WMMA accumulators,
// 1.0 b128 loads per WMMA).
// MODE 0: fp32 out[row*1024 + n] + bias[n]
// MODE 1: bf16 out in [B,H,T,D]       (Q, K)
// MODE 2: bf16 out in [B,H,D,T]       (V transposed for flash P*V)
// ---------------------------------------------------------------------------
template <int MODE>
__global__ __launch_bounds__(32)
void gemm_bf16(const bf16_t* __restrict__ X, const bf16_t* __restrict__ W,
               float* __restrict__ outF, bf16_t* __restrict__ outB,
               const float* __restrict__ bias) {
    const int lane    = threadIdx.x;
    const int nBase   = blockIdx.x * 64;
    const int rowBase = blockIdx.y * 64;

    v8f acc[4][4];
#pragma unroll
    for (int i = 0; i < 4; ++i)
#pragma unroll
        for (int j = 0; j < 4; ++j) acc[i][j] = zero8();

    for (int k = 0; k < D_MODEL; k += 32) {
        v16bf a[4];
#pragma unroll
        for (int mi = 0; mi < 4; ++mi)
            a[mi] = load_tile(X + (size_t)(rowBase + mi * 16) * D_MODEL + k,
                              D_MODEL, lane);
#pragma unroll
        for (int ni = 0; ni < 4; ++ni) {
            v16bf b = load_tile(W + (size_t)(nBase + ni * 16) * D_MODEL + k,
                                D_MODEL, lane);
#pragma unroll
            for (int mi = 0; mi < 4; ++mi)
                acc[mi][ni] = wmma_bf16(a[mi], b, acc[mi][ni]);
        }
    }

    const int mOff = (lane < 16) ? 0 : 8;
    const int nl   = lane & 15;
#pragma unroll
    for (int mi = 0; mi < 4; ++mi) {
#pragma unroll
        for (int ni = 0; ni < 4; ++ni) {
#pragma unroll
            for (int r = 0; r < 8; ++r) {
                const int row = rowBase + mi * 16 + r + mOff;
                const int col = nBase + ni * 16 + nl;
                const float v = acc[mi][ni][r];
                if (MODE == 0) {
                    outF[(size_t)row * D_MODEL + col] = v + bias[col];
                } else {
                    const int b = row >> 11;        // row / 2048
                    const int t = row & 2047;
                    const int h = col >> 6;         // col / 64
                    const int d = col & 63;
                    if (MODE == 1) {
                        outB[(((size_t)(b * NHEADS + h) * SEQ + t) << 6) + d] =
                            (bf16_t)v;
                    } else {
                        outB[((size_t)(b * NHEADS + h) * HDIM + d) * SEQ + t] =
                            (bf16_t)v;
                    }
                }
            }
        }
    }
}

// ---------------------------------------------------------------------------
// RoPE (in place) on Q and K stored [B,H,T,D]; folds 1/sqrt(64) into Q.
// One thread per (bh, t, d2) pair with d2 in [0,32); rotates (d2, d2+32).
// ---------------------------------------------------------------------------
__global__ void rope_kernel(bf16_t* __restrict__ q, bf16_t* __restrict__ k) {
    const int i = blockIdx.x * blockDim.x + threadIdx.x;
    if (i >= BATCH * NHEADS * SEQ * 32) return;
    const int d2 = i & 31;
    const int t  = (i >> 5) & (SEQ - 1);
    const int bh = i >> 16;                       // 32*2048 = 2^16

    // inv_freq = 10000^(-d2/32) ; angle = t * inv_freq
    const float inv_freq = exp2f(-(float)d2 * (13.2877123795494f / 32.0f));
    const float ang = (float)t * inv_freq;
    const float s = __sinf(ang);
    const float c = __cosf(ang);

    const size_t base = ((size_t)bh * SEQ + t) * HDIM + d2;
    {   // Q with 0.125 scale folded in
        const float x1 = (float)q[base];
        const float x2 = (float)q[base + 32];
        q[base]      = (bf16_t)(0.125f * (x1 * c - x2 * s));
        q[base + 32] = (bf16_t)(0.125f * (x2 * c + x1 * s));
    }
    {   // K unscaled
        const float x1 = (float)k[base];
        const float x2 = (float)k[base + 32];
        k[base]      = (bf16_t)(x1 * c - x2 * s);
        k[base + 32] = (bf16_t)(x2 * c + x1 * s);
    }
}

// ---------------------------------------------------------------------------
// Flash attention: one wave handles 32 q-rows (two 16-row tiles) for one
// (b,h), so each K/V tile load feeds two WMMAs (16 b128 loads : 16 WMMAs).
// Softmax strategy: transpose raw scores S through LDS into A-operand order,
// so each lane owns 16 row elements -> per-lane max/sum + ONE partner
// shuffle (no ds_bpermute chains); exp2 output is already the P A-operand.
// Per-row alpha / l go back to the C/D-layout accumulators via a tiny LDS
// broadcast.
// Q,K: [B,H,T,64] bf16 (Q pre-scaled).  VT: [B,H,64,T] bf16.
// O: [B,T,1024] bf16 (heads merged) -> feeds output projection GEMM.
// ---------------------------------------------------------------------------
__global__ __launch_bounds__(32)
void flash_attn(const bf16_t* __restrict__ Q, const bf16_t* __restrict__ K,
                const bf16_t* __restrict__ VT, bf16_t* __restrict__ O) {
    __shared__ float ldsP[32 * 40];               // two 16x32 S tiles, padded
    __shared__ float ldsA[32];                    // per-row alpha / l bcast
    const int lane = threadIdx.x;
    const int tq   = blockIdx.x;                  // 0..63 (32 q-rows each)
    const int bh   = blockIdx.y;                  // 0..63

    const int m    = lane & 15;
    const int mOff = (lane < 16) ? 0 : 8;
    const int nl   = lane & 15;
    const int cb   = (lane < 16) ? 0 : 8;

    const bf16_t* qp = Q + ((size_t)bh * SEQ + tq * 32) * HDIM;
    v16bf aQ[2][2];
#pragma unroll
    for (int qt = 0; qt < 2; ++qt) {
        aQ[qt][0] = load_tile(qp + (size_t)qt * 16 * HDIM,      HDIM, lane);
        aQ[qt][1] = load_tile(qp + (size_t)qt * 16 * HDIM + 32, HDIM, lane);
    }

    v8f acc[2][4];
#pragma unroll
    for (int i = 0; i < 2; ++i)
#pragma unroll
        for (int j = 0; j < 4; ++j) acc[i][j] = zero8();
    float mr[2] = {-1e30f, -1e30f};
    float lr[2] = {0.0f, 0.0f};

    for (int kb = 0; kb < SEQ / 32; ++kb) {
        const bf16_t* kp = K + ((size_t)bh * SEQ + kb * 32) * HDIM;
        // ---- S = Q K^T for 32 keys (two 16-key B tiles, shared by both
        //      q tiles) ----
        const v16bf b0a = load_tile(kp,                  HDIM, lane);
        const v16bf b0b = load_tile(kp + 32,             HDIM, lane);
        const v16bf b1a = load_tile(kp + 16 * HDIM,      HDIM, lane);
        const v16bf b1b = load_tile(kp + 16 * HDIM + 32, HDIM, lane);
        v8f s[2][2];
#pragma unroll
        for (int qt = 0; qt < 2; ++qt) {
            s[qt][0] = wmma_bf16(aQ[qt][0], b0a, zero8());
            s[qt][0] = wmma_bf16(aQ[qt][1], b0b, s[qt][0]);
            s[qt][1] = wmma_bf16(aQ[qt][0], b1a, zero8());
            s[qt][1] = wmma_bf16(aQ[qt][1], b1b, s[qt][1]);
        }

        // ---- transpose S (C/D layout -> A layout) through LDS ----
#pragma unroll
        for (int qt = 0; qt < 2; ++qt)
#pragma unroll
            for (int r = 0; r < 8; ++r) {
                ldsP[(qt * 16 + r + mOff) * 40 + nl]      = s[qt][0][r];
                ldsP[(qt * 16 + r + mOff) * 40 + 16 + nl] = s[qt][1][r];
            }
        asm volatile("s_wait_dscnt 0" ::: "memory");

        // ---- per-lane softmax over one row's 16 elements + 1 shuffle ----
        v16bf aP[2];
#pragma unroll
        for (int qt = 0; qt < 2; ++qt) {
            const int base = (qt * 16 + m) * 40 + cb;
            float v[16];
#pragma unroll
            for (int j = 0; j < 8; ++j) {
                v[j]     = ldsP[base + j];
                v[8 + j] = ldsP[base + 16 + j];
            }
            float bm = v[0];
#pragma unroll
            for (int j = 1; j < 16; ++j) bm = fmaxf(bm, v[j]);
            bm = fmaxf(bm, __shfl_xor(bm, 16, 32));     // join partner lane
            const float mn = fmaxf(mr[qt], bm);
            const float al = exp2f((mr[qt] - mn) * LOG2E);
            float rs = 0.0f;
#pragma unroll
            for (int j = 0; j < 16; ++j) {
                const float p = exp2f((v[j] - mn) * LOG2E);
                aP[qt][j] = (bf16_t)p;                  // already A layout!
                rs += p;
            }
            rs += __shfl_xor(rs, 16, 32);
            lr[qt] = lr[qt] * al + rs;
            mr[qt] = mn;
            ldsA[qt * 16 + m] = al;                     // broadcast alpha
        }
        asm volatile("s_wait_dscnt 0" ::: "memory");

        // ---- rescale accumulators with per-row alpha ----
#pragma unroll
        for (int qt = 0; qt < 2; ++qt)
#pragma unroll
            for (int r = 0; r < 8; ++r) {
                const float al = ldsA[qt * 16 + r + mOff];
#pragma unroll
                for (int dt = 0; dt < 4; ++dt) acc[qt][dt][r] *= al;
            }

        // ---- O += P * V  (VT gives contiguous keys per lane; V tiles
        //      shared by both q tiles) ----
        const bf16_t* vp = VT + (size_t)bh * HDIM * SEQ + kb * 32;
#pragma unroll
        for (int dt = 0; dt < 4; ++dt) {
            v16bf bv = load_tile(vp + (size_t)(dt * 16) * SEQ, SEQ, lane);
#pragma unroll
            for (int qt = 0; qt < 2; ++qt)
                acc[qt][dt] = wmma_bf16(aP[qt], bv, acc[qt][dt]);
        }
    }

    // ---- epilogue: normalize, merge heads, store bf16 ----
#pragma unroll
    for (int qt = 0; qt < 2; ++qt) ldsA[qt * 16 + m] = lr[qt];
    asm volatile("s_wait_dscnt 0" ::: "memory");

    const int b = bh >> 4;
    const int h = bh & 15;
#pragma unroll
    for (int qt = 0; qt < 2; ++qt) {
#pragma unroll
        for (int r = 0; r < 8; ++r) {
            const float linv = 1.0f / ldsA[qt * 16 + r + mOff];
            const int t = tq * 32 + qt * 16 + r + mOff;
#pragma unroll
            for (int dt = 0; dt < 4; ++dt) {
                const int col = h * HDIM + dt * 16 + nl;
                O[((size_t)b * SEQ + t) * D_MODEL + col] =
                    (bf16_t)(acc[qt][dt][r] * linv);
            }
        }
    }
}

// ---------------------------------------------------------------------------
// Host-side launcher
// ---------------------------------------------------------------------------
extern "C" void kernel_launch(void* const* d_in, const int* in_sizes, int n_in,
                              void* d_out, int out_size, void* d_ws,
                              size_t ws_size, hipStream_t stream) {
    const float* x  = (const float*)d_in[0];
    const float* Wq = (const float*)d_in[1];
    const float* Wk = (const float*)d_in[2];
    const float* Wv = (const float*)d_in[3];
    const float* Wo = (const float*)d_in[4];
    const float* bo = (const float*)d_in[5];
    float* out = (float*)d_out;

    // Workspace carve-up (~88 MiB total)
    char* w = (char*)d_ws;
    const size_t xElems = (size_t)ROWS * D_MODEL;       // 8.39M
    const size_t wElems = (size_t)D_MODEL * D_MODEL;    // 1.05M
    bf16_t* xb  = (bf16_t*)w;  w += xElems * 2;         // 16 MiB
    bf16_t* wqb = (bf16_t*)w;  w += wElems * 2;         //  2 MiB
    bf16_t* wkb = (bf16_t*)w;  w += wElems * 2;
    bf16_t* wvb = (bf16_t*)w;  w += wElems * 2;
    bf16_t* wob = (bf16_t*)w;  w += wElems * 2;
    bf16_t* qb  = (bf16_t*)w;  w += xElems * 2;         // [B,H,T,D]
    bf16_t* kbf = (bf16_t*)w;  w += xElems * 2;         // [B,H,T,D]
    bf16_t* vtb = (bf16_t*)w;  w += xElems * 2;         // [B,H,D,T]
    bf16_t* ob  = (bf16_t*)w;  w += xElems * 2;         // [B,T,1024]

    // 1) fp32 -> bf16
    cvt_f32_bf16<<<(int)((xElems + 255) / 256), 256, 0, stream>>>(x, xb, (int)xElems);
    cvt_f32_bf16<<<(int)((wElems + 255) / 256), 256, 0, stream>>>(Wq, wqb, (int)wElems);
    cvt_f32_bf16<<<(int)((wElems + 255) / 256), 256, 0, stream>>>(Wk, wkb, (int)wElems);
    cvt_f32_bf16<<<(int)((wElems + 255) / 256), 256, 0, stream>>>(Wv, wvb, (int)wElems);
    cvt_f32_bf16<<<(int)((wElems + 255) / 256), 256, 0, stream>>>(Wo, wob, (int)wElems);

    // 2) projections (WMMA): grid = (N/64, ROWS/64), 1 wave per block
    dim3 ggrid(D_MODEL / 64, ROWS / 64);
    gemm_bf16<1><<<ggrid, 32, 0, stream>>>(xb, wqb, nullptr, qb, nullptr);
    gemm_bf16<1><<<ggrid, 32, 0, stream>>>(xb, wkb, nullptr, kbf, nullptr);
    gemm_bf16<2><<<ggrid, 32, 0, stream>>>(xb, wvb, nullptr, vtb, nullptr);

    // 3) RoPE on Q,K (folds softmax scale into Q)
    {
        const int n = BATCH * NHEADS * SEQ * 32;
        rope_kernel<<<(n + 255) / 256, 256, 0, stream>>>(qb, kbf);
    }

    // 4) flash attention: grid = (T/32 q-tiles, B*H)
    flash_attn<<<dim3(SEQ / 32, BATCH * NHEADS), 32, 0, stream>>>(qb, kbf, vtb, ob);

    // 5) output projection + bias (fp32 result)
    gemm_bf16<0><<<ggrid, 32, 0, stream>>>(ob, wob, out, nullptr, bo);
}
